// DecoderRNNT_55722905698591
// MI455X (gfx1250) — matrix operand order repaired and checked by reading, not compiled, not run on hardware
//
#include <hip/hip_runtime.h>
#include <hip/hip_bf16.h>

#define B_   8
#define T_   200
#define U_   50
#define DU   512      // DUNITS == EMB == EPROJS == JOINT
#define OD   1000     // ODIM
#define ODP  1008     // ODIM padded to multiple of 16
#define G4   2048     // 4*DUNITS (LSTM gates)
#define NT_OUT (ODP / 16)   // 63 N-tiles in the output GEMM

typedef __attribute__((ext_vector_type(16))) _Float16 v16h;
typedef __attribute__((ext_vector_type(8)))  float    v8f;

__device__ __forceinline__ float sigmoidf_(float x) {
    return 1.0f / (1.0f + __expf(-x));
}

__device__ __forceinline__ v8f wmma_f16(v16h a, v16h b, v8f c) {
    // D = A(16x32 f16) * B(32x16 f16) + C(16x16 f32)
    return __builtin_amdgcn_wmma_f32_16x16x32_f16(
        /*neg_a=*/false, a, /*neg_b=*/false, b,
        /*c_mod=*/(short)0, c, /*reuse_a=*/false, /*reuse_b=*/false);
}

// ---- A fragment: 16x32 (MxK) from row-major src, row stride ld halves.
// Lanes 0-15 hold K=0..15, lanes 16-31 hold K=16..31 (matches B fragment).
// All 16 rows must be valid (pad with zeros in src).
__device__ __forceinline__ v16h load_a_frag(const _Float16* __restrict__ src,
                                            int ld, int k0, int lane) {
    const int m  = lane & 15;
    const int kh = (lane >> 4) & 1;
    const _Float16* p = src + (size_t)m * ld + k0 + kh * 16;
    v16h a;
#pragma unroll
    for (int i = 0; i < 8; ++i) a[i] = p[i];
    p += 8;
#pragma unroll
    for (int i = 0; i < 8; ++i) a[8 + i] = p[i];
    return a;
}

// ---- B fragment: 32x16 (KxN) where B[k][n] = W[n0+n][k0+k], W row-major NxK.
// Contiguous k order, same as A (not the ISA table order): lanes 0-15 hold K=0..15, lanes 16-31 hold K=16..31,
// i.e. 16 contiguous halves per lane (2x b128 loads).
__device__ __forceinline__ v16h load_b_frag(const _Float16* __restrict__ W,
                                            int ldk, int n0, int k0, int lane) {
    const int n  = lane & 15;
    const int kh = (lane >> 4) & 1;
    const _Float16* p = W + (size_t)(n0 + n) * ldk + k0 + kh * 16;
    v16h b;
#pragma unroll
    for (int i = 0; i < 16; ++i) b[i] = p[i];
    return b;
}

// ============================ prep kernels ============================

__global__ void cvt_f16(const float* __restrict__ s, _Float16* __restrict__ d, int n) {
    int i = blockIdx.x * blockDim.x + threadIdx.x;
    if (i < n) d[i] = (_Float16)s[i];
}

__global__ void cvt_wout(const float* __restrict__ s, _Float16* __restrict__ d) {
    int i = blockIdx.x * blockDim.x + threadIdx.x;   // ODP*DU
    if (i < ODP * DU) {
        int row = i >> 9;                            // /512
        d[i] = (row < OD) ? (_Float16)s[i] : (_Float16)0.0f;
    }
}

__global__ void embed_k(const int* __restrict__ ys, const float* __restrict__ emb,
                        _Float16* __restrict__ eys) {
    int i = blockIdx.x * blockDim.x + threadIdx.x;   // B*U*DU
    if (i < B_ * U_ * DU) {
        int bu = i >> 9, k = i & (DU - 1);
        int tok = ys[bu];                            // BLANK==0 -> zero row
        eys[i] = (tok == 0) ? (_Float16)0.0f : (_Float16)emb[(size_t)tok * DU + k];
    }
}

// ============================ LSTM scan ============================
// One persistent workgroup (16 waves). Per step, per layer:
//   gates(8x2048) = x(8x512)@W_ih^T + h(8x512)@W_hh^T   via WMMA M-tile of 16.
// kt-outer / acc[8] structure: one A-fragment feeds 8 WMMAs (one per N-tile).
// x is staged into LDS with the CDNA5 async-to-LDS path (ASYNCcnt-tracked,
// no VGPR round trip). Weights (8MB f16) are L2-resident across all 50 steps.
__global__ void __launch_bounds__(512)
lstm_scan(const _Float16* __restrict__ eys,
          const _Float16* __restrict__ Wih0, const _Float16* __restrict__ Whh0,
          const float* __restrict__ bih0,   const float* __restrict__ bhh0,
          const _Float16* __restrict__ Wih1, const _Float16* __restrict__ Whh1,
          const float* __restrict__ bih1,   const float* __restrict__ bhh1,
          _Float16* __restrict__ h_dec)
{
    __shared__ _Float16 xbuf[16 * DU];    // 16KB (rows 8..15 zero)
    __shared__ _Float16 h0[16 * DU];      // 16KB (rows 8..15 zero)
    __shared__ _Float16 h1[16 * DU];      // 16KB
    __shared__ float    c0[B_ * DU];      // 16KB
    __shared__ float    c1[B_ * DU];      // 16KB
    __shared__ float    gates[B_ * G4];   // 64KB
    __shared__ float    bsum0[G4];        // 8KB
    __shared__ float    bsum1[G4];        // 8KB   total 160KB < 320KB/WGP

    const int tid  = threadIdx.x;
    const int lane = tid & 31;
    const int wave = tid >> 5;            // 0..15
    const int n    = lane & 15;
    const int mhi  = (lane >> 4) & 1;

    for (int i = tid; i < 16 * DU; i += 512) {
        xbuf[i] = (_Float16)0.0f; h0[i] = (_Float16)0.0f; h1[i] = (_Float16)0.0f;
    }
    for (int i = tid; i < B_ * DU; i += 512) { c0[i] = 0.0f; c1[i] = 0.0f; }
    for (int i = tid; i < G4; i += 512) { bsum0[i] = bih0[i] + bhh0[i]; bsum1[i] = bih1[i] + bhh1[i]; }
    __syncthreads();

    // LDS byte offset of xbuf: generic LDS addresses are aperture_base|offset
    // with the aperture base's low 32 bits zero, so truncation yields the offset.
    const unsigned xbase = (unsigned)(size_t)(void*)xbuf;
    // each of the 512 threads async-copies one 16B chunk: 8 rows x 1KB = 8KB
    const unsigned obyte  = (unsigned)tid * 16u;
    const int      bb_cp  = obyte >> 10;               // row (batch) 0..7
    const unsigned jb_cp  = obyte & 1023u;             // byte within row
    const unsigned ldsdst = xbase + obyte;

    for (int u = 0; u < U_; ++u) {
        // ---- async stage x = eys[:,u,:] into LDS (rows 8..15 stay zero) ----
        {
            const char* gsrc = (const char*)eys
                             + ((size_t)bb_cp * U_ + u) * (DU * 2) + jb_cp;
            asm volatile("global_load_async_to_lds_b128 %0, %1, off"
                         :: "v"(ldsdst), "v"(gsrc) : "memory");
            asm volatile("s_wait_asynccnt 0x0" ::: "memory");
        }
        __syncthreads();

        // ---------- layer 0 gates ----------
        {
            v8f acc[8];
#pragma unroll
            for (int t = 0; t < 8; ++t) acc[t] = (v8f){};
#pragma unroll 1
            for (int kt = 0; kt < 16; ++kt) {
                v16h a = load_a_frag(xbuf, DU, kt * 32, lane);
#pragma unroll
                for (int t = 0; t < 8; ++t) {
                    v16h b = load_b_frag(Wih0, DU, (wave * 8 + t) * 16, kt * 32, lane);
                    acc[t] = wmma_f16(a, b, acc[t]);
                }
            }
#pragma unroll 1
            for (int kt = 0; kt < 16; ++kt) {
                v16h a = load_a_frag(h0, DU, kt * 32, lane);
#pragma unroll
                for (int t = 0; t < 8; ++t) {
                    v16h b = load_b_frag(Whh0, DU, (wave * 8 + t) * 16, kt * 32, lane);
                    acc[t] = wmma_f16(a, b, acc[t]);
                }
            }
#pragma unroll
            for (int t = 0; t < 8; ++t) {
                const int n0 = (wave * 8 + t) * 16;
#pragma unroll
                for (int r = 0; r < 8; ++r) {
                    const int m = mhi * 8 + r;
                    if (m < B_) gates[m * G4 + n0 + n] = acc[t][r];
                }
            }
        }
        __syncthreads();
#pragma unroll 1
        for (int e = tid; e < B_ * DU; e += 512) {
            const int bb = e >> 9, j = e & (DU - 1);
            const float gi = gates[bb * G4 + j]          + bsum0[j];
            const float gf = gates[bb * G4 + DU + j]     + bsum0[DU + j];
            const float gg = gates[bb * G4 + 2 * DU + j] + bsum0[2 * DU + j];
            const float go = gates[bb * G4 + 3 * DU + j] + bsum0[3 * DU + j];
            const float c  = sigmoidf_(gf) * c0[e] + sigmoidf_(gi) * tanhf(gg);
            c0[e] = c;
            h0[bb * DU + j] = (_Float16)(sigmoidf_(go) * tanhf(c));
        }
        __syncthreads();

        // ---------- layer 1 gates: x = h0, hidden h1/c1 ----------
        {
            v8f acc[8];
#pragma unroll
            for (int t = 0; t < 8; ++t) acc[t] = (v8f){};
#pragma unroll 1
            for (int kt = 0; kt < 16; ++kt) {
                v16h a = load_a_frag(h0, DU, kt * 32, lane);
#pragma unroll
                for (int t = 0; t < 8; ++t) {
                    v16h b = load_b_frag(Wih1, DU, (wave * 8 + t) * 16, kt * 32, lane);
                    acc[t] = wmma_f16(a, b, acc[t]);
                }
            }
#pragma unroll 1
            for (int kt = 0; kt < 16; ++kt) {
                v16h a = load_a_frag(h1, DU, kt * 32, lane);
#pragma unroll
                for (int t = 0; t < 8; ++t) {
                    v16h b = load_b_frag(Whh1, DU, (wave * 8 + t) * 16, kt * 32, lane);
                    acc[t] = wmma_f16(a, b, acc[t]);
                }
            }
#pragma unroll
            for (int t = 0; t < 8; ++t) {
                const int n0 = (wave * 8 + t) * 16;
#pragma unroll
                for (int r = 0; r < 8; ++r) {
                    const int m = mhi * 8 + r;
                    if (m < B_) gates[m * G4 + n0 + n] = acc[t][r];
                }
            }
        }
        __syncthreads();
#pragma unroll 1
        for (int e = tid; e < B_ * DU; e += 512) {
            const int bb = e >> 9, j = e & (DU - 1);
            const float gi = gates[bb * G4 + j]          + bsum1[j];
            const float gf = gates[bb * G4 + DU + j]     + bsum1[DU + j];
            const float gg = gates[bb * G4 + 2 * DU + j] + bsum1[2 * DU + j];
            const float go = gates[bb * G4 + 3 * DU + j] + bsum1[3 * DU + j];
            const float c  = sigmoidf_(gf) * c1[e] + sigmoidf_(gi) * tanhf(gg);
            c1[e] = c;
            const float h = sigmoidf_(go) * tanhf(c);
            h1[bb * DU + j] = (_Float16)h;
            h_dec[((size_t)bb * U_ + u) * DU + j] = (_Float16)h;
        }
        __syncthreads();
    }
}

// ============================ Z = X @ W^T (+bias), N=K=512 ============================
__global__ void __launch_bounds__(256)
gemm_xt(const _Float16* __restrict__ X, const _Float16* __restrict__ W,
        const float* __restrict__ bias, float* __restrict__ Z)
{
    const int lane = threadIdx.x & 31;
    const int wave = threadIdx.x >> 5;                   // 0..7
    const int row0 = blockIdx.x * 16;
    const int n0   = (blockIdx.y * 8 + wave) * 16;
    const int n    = lane & 15;
    const int mhi  = (lane >> 4) & 1;

    const float bv = bias ? bias[n0 + n] : 0.0f;
    v8f acc;
#pragma unroll
    for (int r = 0; r < 8; ++r) acc[r] = bv;

#pragma unroll 1
    for (int kt = 0; kt < 16; ++kt) {
        v16h a = load_a_frag(X + (size_t)row0 * DU, DU, kt * 32, lane);
        v16h b = load_b_frag(W, DU, n0, kt * 32, lane);
        acc = wmma_f16(a, b, acc);
    }
#pragma unroll
    for (int r = 0; r < 8; ++r) {
        const int m = mhi * 8 + r;
        Z[(size_t)(row0 + m) * DU + n0 + n] = acc[r];
    }
}

// ============================ fused joint + output GEMM ============================
// One block per (b,t): all 64 u-rows (50 valid) staged as tanh(z_enc+z_dec) f16
// in LDS, then each wave sweeps its N-tiles of W_out with each B-fragment
// reused across 4 M-subtiles (acc[4]) -> 4x less W_out traffic from L2.
__global__ void __launch_bounds__(256)
joint_out(const float* __restrict__ zenc, const float* __restrict__ zdec,
          const _Float16* __restrict__ WoutP, const float* __restrict__ bout,
          float* __restrict__ out)
{
    __shared__ _Float16 Atile[64 * DU];                  // 64KB

    const int bt   = blockIdx.x;                         // 0..B*T-1
    const int b    = bt / T_;
    const int lane = threadIdx.x & 31;
    const int wave = threadIdx.x >> 5;
    const int n    = lane & 15;
    const int mhi  = (lane >> 4) & 1;

    const float* ze = zenc + (size_t)bt * DU;
    for (int e = threadIdx.x; e < 64 * DU; e += 256) {
        const int u = e >> 9, k = e & (DU - 1);
        float v = 0.0f;
        if (u < U_) v = tanhf(ze[k] + zdec[((size_t)b * U_ + u) * DU + k]);
        Atile[e] = (_Float16)v;
    }
    __syncthreads();

#pragma unroll 1
    for (int i = 0; i < 8; ++i) {
        const int nt = wave + 8 * i;                     // interleaved N-tiles
        if (nt >= NT_OUT) break;                         // only wave 7, i 7
        const int n0 = nt * 16;
        const int ng = n0 + n;
        const float bv = (ng < OD) ? bout[ng] : 0.0f;

        v8f acc[4];
#pragma unroll
        for (int mi = 0; mi < 4; ++mi)
#pragma unroll
            for (int r = 0; r < 8; ++r) acc[mi][r] = bv;

#pragma unroll 1
        for (int kt = 0; kt < 16; ++kt) {
            v16h bf = load_b_frag(WoutP, DU, n0, kt * 32, lane);
#pragma unroll
            for (int mi = 0; mi < 4; ++mi) {
                v16h a = load_a_frag(Atile + mi * 16 * DU, DU, kt * 32, lane);
                acc[mi] = wmma_f16(a, bf, acc[mi]);
            }
        }

        if (ng < OD) {
#pragma unroll
            for (int mi = 0; mi < 4; ++mi)
#pragma unroll
                for (int r = 0; r < 8; ++r) {
                    const int u = mi * 16 + mhi * 8 + r;
                    if (u < U_)
                        out[((size_t)bt * U_ + u) * OD + ng] = acc[mi][r];
                }
        }
    }
}

// ============================ launch ============================

extern "C" void kernel_launch(void* const* d_in, const int* in_sizes, int n_in,
                              void* d_out, int out_size, void* d_ws, size_t ws_size,
                              hipStream_t stream) {
    (void)in_sizes; (void)n_in; (void)out_size; (void)ws_size;

    const float* hs    = (const float*)d_in[0];   // (B,T,512)
    const int*   ys    = (const int*)  d_in[1];   // (B,U)
    const float* emb   = (const float*)d_in[2];   // (1000,512)
    const float* Wih0  = (const float*)d_in[3];
    const float* Whh0  = (const float*)d_in[4];
    const float* bih0  = (const float*)d_in[5];
    const float* bhh0  = (const float*)d_in[6];
    const float* Wih1  = (const float*)d_in[7];
    const float* Whh1  = (const float*)d_in[8];
    const float* bih1  = (const float*)d_in[9];
    const float* bhh1  = (const float*)d_in[10];
    const float* Wenc  = (const float*)d_in[11];
    const float* benc  = (const float*)d_in[12];
    const float* Wdec  = (const float*)d_in[13];
    const float* Wout  = (const float*)d_in[14];
    const float* bout  = (const float*)d_in[15];

    char* ws = (char*)d_ws;
    size_t off = 0;
    auto alloc = [&](size_t bytes) -> char* {
        char* p = ws + off;
        off = (off + bytes + 255) & ~(size_t)255;
        return p;
    };
    _Float16* eys16  = (_Float16*)alloc((size_t)B_ * U_ * DU * 2);
    _Float16* hs16   = (_Float16*)alloc((size_t)B_ * T_ * DU * 2);
    _Float16* Wih0h  = (_Float16*)alloc((size_t)G4 * DU * 2);
    _Float16* Whh0h  = (_Float16*)alloc((size_t)G4 * DU * 2);
    _Float16* Wih1h  = (_Float16*)alloc((size_t)G4 * DU * 2);
    _Float16* Whh1h  = (_Float16*)alloc((size_t)G4 * DU * 2);
    _Float16* Wench  = (_Float16*)alloc((size_t)DU * DU * 2);
    _Float16* Wdech  = (_Float16*)alloc((size_t)DU * DU * 2);
    _Float16* Wouth  = (_Float16*)alloc((size_t)ODP * DU * 2);
    _Float16* hdec16 = (_Float16*)alloc((size_t)B_ * U_ * DU * 2);
    float*    zenc   = (float*)   alloc((size_t)B_ * T_ * DU * 4);
    float*    zdec   = (float*)   alloc((size_t)B_ * U_ * DU * 4);

    auto blocks = [](int n) { return (n + 255) / 256; };

    // ---- convert weights/activations to f16 ----
    cvt_f16<<<blocks(B_ * T_ * DU), 256, 0, stream>>>(hs,   hs16,  B_ * T_ * DU);
    cvt_f16<<<blocks(G4 * DU),      256, 0, stream>>>(Wih0, Wih0h, G4 * DU);
    cvt_f16<<<blocks(G4 * DU),      256, 0, stream>>>(Whh0, Whh0h, G4 * DU);
    cvt_f16<<<blocks(G4 * DU),      256, 0, stream>>>(Wih1, Wih1h, G4 * DU);
    cvt_f16<<<blocks(G4 * DU),      256, 0, stream>>>(Whh1, Whh1h, G4 * DU);
    cvt_f16<<<blocks(DU * DU),      256, 0, stream>>>(Wenc, Wench, DU * DU);
    cvt_f16<<<blocks(DU * DU),      256, 0, stream>>>(Wdec, Wdech, DU * DU);
    cvt_wout<<<blocks(ODP * DU),    256, 0, stream>>>(Wout, Wouth);
    embed_k<<<blocks(B_ * U_ * DU), 256, 0, stream>>>(ys, emb, eys16);

    // ---- sequential 2-layer LSTM over U steps (single persistent WGP) ----
    lstm_scan<<<1, 512, 0, stream>>>(eys16, Wih0h, Whh0h, bih0, bhh0,
                                     Wih1h, Whh1h, bih1, bhh1, hdec16);

    // ---- z_enc = hs @ Wenc^T + benc ; z_dec = h_dec @ Wdec^T ----
    gemm_xt<<<dim3((B_ * T_) / 16, 4), 256, 0, stream>>>(hs16,   Wench, benc,    zenc);
    gemm_xt<<<dim3((B_ * U_) / 16, 4), 256, 0, stream>>>(hdec16, Wdech, nullptr, zdec);

    // ---- out = tanh(z_enc[:, :, None] + z_dec[:, None, :]) @ Wout^T + bout ----
    joint_out<<<B_ * T_, 256, 0, stream>>>(zenc, zdec, Wouth, bout, (float*)d_out);
}
